// MoE_35871566856897
// MI455X (gfx1250) — matrix operand order repaired and checked by reading, not compiled, not run on hardware
//
#include <hip/hip_runtime.h>

typedef __attribute__((ext_vector_type(16))) __bf16 v16bf;
typedef __attribute__((ext_vector_type(8)))  __bf16 v8bf;
typedef __attribute__((ext_vector_type(8)))  float  v8f;

#define B_TOK 2048
#define NEXP  128
#define TOPK  8
#define CDIM  256
#define HDIM  512
#define ROWS  64        // N * L2
#define CAPN  20        // ceil(1.25 * 2048 / 128)
#define XS    264       // padded LDS row stride for X tile (bf16 elems)

__device__ __forceinline__ v16bf cat16(v8bf lo, v8bf hi) {
  return __builtin_shufflevector(lo, hi, 0,1,2,3,4,5,6,7,8,9,10,11,12,13,14,15);
}
// A-matrix fragment: lane<16 holds K {0..7, 16..23}, lane>=16 holds {8..15, 24..31}
__device__ __forceinline__ v16bf load_frag_a(const __bf16* p) {
  v8bf lo = *(const v8bf*)p;
  v8bf hi = *(const v8bf*)(p + 16);
  return cat16(lo, hi);
}
// B-matrix fragment: lane<16 holds K {0..7, 16..23}, lane>=16 holds {8..15, 24..31} of one N column
__device__ __forceinline__ v16bf load_frag_b(const __bf16* p) {
  v8bf lo = *(const v8bf*)p;
  v8bf hi = *(const v8bf*)(p + 16);
  return cat16(lo, hi);
}

// ---------------- router: logits -> softmax -> top-8 (stable) ----------------
__global__ __launch_bounds__(128) void router_kernel(
    const float* __restrict__ fea, const float* __restrict__ rw,
    float* __restrict__ topk_w, int* __restrict__ flat_exp,
    float* __restrict__ importance) {
  const int b = blockIdx.x, t = threadIdx.x;
  __shared__ float sp[128];
  __shared__ float sv[128];
  __shared__ int   si[128];
  __shared__ float tval[TOPK];
  __shared__ int   tidx[TOPK];

  const float* f = fea + (size_t)b * CDIM;
  float dot = 0.0f;
  for (int c = 0; c < CDIM; ++c) dot += f[c] * rw[c * NEXP + t];

  sv[t] = dot; __syncthreads();
  for (int o = 64; o >= 1; o >>= 1) { if (t < o) sv[t] = fmaxf(sv[t], sv[t + o]); __syncthreads(); }
  const float mx = sv[0]; __syncthreads();
  const float ex = __expf(dot - mx);
  sv[t] = ex; __syncthreads();
  for (int o = 64; o >= 1; o >>= 1) { if (t < o) sv[t] += sv[t + o]; __syncthreads(); }
  const float prob = ex / sv[0];
  __syncthreads();
  atomicAdd(&importance[t], prob);
  sp[t] = prob; __syncthreads();

  for (int k = 0; k < TOPK; ++k) {
    sv[t] = sp[t]; si[t] = t; __syncthreads();
    for (int o = 64; o >= 1; o >>= 1) {
      if (t < o) {
        if (sv[t + o] > sv[t] || (sv[t + o] == sv[t] && si[t + o] < si[t])) {
          sv[t] = sv[t + o]; si[t] = si[t + o];
        }
      }
      __syncthreads();
    }
    if (t == 0) { tval[k] = sv[0]; tidx[k] = si[0]; sp[si[0]] = -1.0f; }
    __syncthreads();
  }
  if (t == 0) {
    float s = 0.0f;
    for (int k = 0; k < TOPK; ++k) s += tval[k];
    const float inv = 1.0f / (s + 1e-9f);
    for (int k = 0; k < TOPK; ++k) {
      topk_w[b * TOPK + k] = tval[k] * inv;
      flat_exp[b * TOPK + k] = tidx[k];
    }
  }
}

// -------- per-expert rank (stable sort by weight desc), capacity, tok_slot ----
__global__ __launch_bounds__(256) void rank_kernel(
    const float* __restrict__ topk_w, const int* __restrict__ flat_exp,
    int* __restrict__ rank, int* __restrict__ tok_slot, float* __restrict__ loadv) {
  const int e = blockIdx.x, tid = threadIdx.x;
  __shared__ int   cnt;
  __shared__ int   lst[B_TOK];
  __shared__ float wl[B_TOK];
  if (tid == 0) cnt = 0;
  __syncthreads();
  for (int i = tid; i < B_TOK * TOPK; i += 256) {
    if (flat_exp[i] == e) {
      int p = atomicAdd(&cnt, 1);
      lst[p] = i; wl[p] = topk_w[i];
    }
  }
  __syncthreads();
  const int n = cnt;
  for (int p = tid; p < n; p += 256) {
    const int i = lst[p];
    const float wi = wl[p];
    int r = 0;
    for (int q = 0; q < n; ++q) {
      const float wj = wl[q];
      r += (wj > wi) || (wj == wi && lst[q] < i);
    }
    rank[i] = r;
    if (r < CAPN) tok_slot[e * CAPN + r] = i >> 3;   // flat_tok = i / K
  }
  if (tid == 0) loadv[e] = (float)(n < CAPN ? n : CAPN);
}

// -------------- per-token renormalize kept weights; fill w_slot --------------
__global__ void renorm_kernel(const float* __restrict__ topk_w,
                              const int* __restrict__ flat_exp,
                              const int* __restrict__ rank,
                              float* __restrict__ w_slot) {
  const int b = blockIdx.x * blockDim.x + threadIdx.x;
  if (b >= B_TOK) return;
  float s = 0.0f;
  for (int k = 0; k < TOPK; ++k) {
    const int i = b * TOPK + k;
    if (rank[i] < CAPN) s += topk_w[i];
  }
  const float inv = 1.0f / (s + 1e-9f);
  for (int k = 0; k < TOPK; ++k) {
    const int i = b * TOPK + k;
    if (rank[i] < CAPN) w_slot[flat_exp[i] * CAPN + rank[i]] = topk_w[i] * inv;
  }
}

// ------------- weight transpose + f32 -> bf16 (w1: [E,C,H]->[E,H,C]) ---------
__global__ void convert_w1(const float* __restrict__ w1, __bf16* __restrict__ w1t) {
  const size_t idx = (size_t)blockIdx.x * 256 + threadIdx.x;   // (e*H + h)*C + c
  const int c = (int)(idx % CDIM);
  const size_t r = idx / CDIM;
  const int h = (int)(r % HDIM);
  const size_t e = r / HDIM;
  w1t[idx] = (__bf16)w1[(e * CDIM + c) * HDIM + h];
}
// w2: [E,H,C] -> [E,C,H]
__global__ void convert_w2(const float* __restrict__ w2, __bf16* __restrict__ w2t) {
  const size_t idx = (size_t)blockIdx.x * 256 + threadIdx.x;   // (e*C + c)*H + h
  const int h = (int)(idx % HDIM);
  const size_t r = idx / HDIM;
  const int c = (int)(r % CDIM);
  const size_t e = r / CDIM;
  w2t[idx] = (__bf16)w2[(e * HDIM + h) * CDIM + c];
}

// ------------- main grouped-MLP kernel: one block per (expert, slot) ---------
__global__ __launch_bounds__(256) void moe_expert_kernel(
    const float* __restrict__ xall, const int* __restrict__ tok_slot,
    const float* __restrict__ w_slot, const __bf16* __restrict__ w1t,
    const __bf16* __restrict__ w2t, const float* __restrict__ b1,
    const float* __restrict__ b2, float* __restrict__ y) {
  __shared__ __align__(16) __bf16 smem[ROWS * HDIM];   // 64 KB, aliased X then h
  const int blk = blockIdx.x;
  const float wv = w_slot[blk];
  if (wv == 0.0f) return;                              // empty/dropped slot: no-op
  const int e   = blk / CAPN;
  const int tok = tok_slot[blk];
  const int tid = threadIdx.x;
  const int wave = tid >> 5;
  const int lane = tid & 31;
  const int lm = lane & 15;
  const int hl = lane >> 4;

  // stage X tile [64,256] f32 -> bf16 LDS (padded stride XS)
  {
    const float4* xs = (const float4*)(xall + (size_t)tok * (ROWS * CDIM));
    for (int i = tid; i < ROWS * CDIM / 4; i += 256) {
      const int row = i >> 6;
      const int c4  = (i & 63) << 2;
      float4 v = xs[i];
      __bf16* d = smem + row * XS + c4;
      d[0] = (__bf16)v.x; d[1] = (__bf16)v.y; d[2] = (__bf16)v.z; d[3] = (__bf16)v.w;
    }
  }
  __syncthreads();

  // ---- GEMM1: h = silu(X @ W1 + b1); wave owns cols [wave*64, wave*64+63]
  v8f vzero = {};
  v8f acc1[4][4];
#pragma unroll
  for (int m = 0; m < 4; ++m)
#pragma unroll
    for (int n = 0; n < 4; ++n) acc1[m][n] = vzero;

  const __bf16* w1e = w1t + (size_t)e * HDIM * CDIM;
#pragma unroll 1
  for (int kk = 0; kk < CDIM; kk += 32) {
    v16bf a[4];
#pragma unroll
    for (int m = 0; m < 4; ++m)
      a[m] = load_frag_a(smem + (m * 16 + lm) * XS + kk + hl * 8);
#pragma unroll
    for (int n = 0; n < 4; ++n) {
      const int col = wave * 64 + n * 16 + lm;
      v16bf bfr = load_frag_b(w1e + (size_t)col * CDIM + kk + hl * 8);
#pragma unroll
      for (int m = 0; m < 4; ++m)
        acc1[m][n] = __builtin_amdgcn_wmma_f32_16x16x32_bf16(
            false, a[m], false, bfr, (short)0, acc1[m][n], false, false);
    }
  }
  __syncthreads();   // all waves done reading X; safe to overwrite with h

  // bias + SiLU, store h row-major [64][512] bf16
#pragma unroll
  for (int n = 0; n < 4; ++n) {
    const int col = wave * 64 + n * 16 + lm;
    const float bias = b1[e * HDIM + col];
#pragma unroll
    for (int m = 0; m < 4; ++m) {
#pragma unroll
      for (int j = 0; j < 8; ++j) {
        float v = acc1[m][n][j] + bias;
        v = v / (1.0f + __expf(-v));
        smem[(m * 16 + j + hl * 8) * HDIM + col] = (__bf16)v;
      }
    }
  }
  __syncthreads();

  // ---- GEMM2: out = h @ W2 + b2; wave owns cols [wave*32, wave*32+31]
  v8f acc2[4][2];
#pragma unroll
  for (int m = 0; m < 4; ++m) { acc2[m][0] = vzero; acc2[m][1] = vzero; }
  const __bf16* w2e = w2t + (size_t)e * CDIM * HDIM;
#pragma unroll 1
  for (int kk = 0; kk < HDIM; kk += 32) {
    v16bf a[4];
#pragma unroll
    for (int m = 0; m < 4; ++m)
      a[m] = load_frag_a(smem + (m * 16 + lm) * HDIM + kk + hl * 8);
#pragma unroll
    for (int n = 0; n < 2; ++n) {
      const int col = wave * 32 + n * 16 + lm;
      v16bf bfr = load_frag_b(w2e + (size_t)col * HDIM + kk + hl * 8);
#pragma unroll
      for (int m = 0; m < 4; ++m)
        acc2[m][n] = __builtin_amdgcn_wmma_f32_16x16x32_bf16(
            false, a[m], false, bfr, (short)0, acc2[m][n], false, false);
    }
  }

  // weighted scatter-add back to tokens
  float* yb = y + (size_t)tok * (ROWS * CDIM);
#pragma unroll
  for (int n = 0; n < 2; ++n) {
    const int col = wave * 32 + n * 16 + lm;
    const float bias = b2[e * CDIM + col];
#pragma unroll
    for (int m = 0; m < 4; ++m) {
#pragma unroll
      for (int j = 0; j < 8; ++j) {
        const int row = m * 16 + j + hl * 8;
        atomicAdd(yb + row * CDIM + col, wv * (acc2[m][n][j] + bias));
      }
    }
  }
}

// ------------------------------- aux loss ------------------------------------
__global__ __launch_bounds__(128) void aux_kernel(const float* __restrict__ imp,
                                                  const float* __restrict__ ld,
                                                  float* __restrict__ out) {
  __shared__ float s1[128], s2[128];
  const int t = threadIdx.x;
  s1[t] = imp[t]; s2[t] = ld[t]; __syncthreads();
  for (int o = 64; o >= 1; o >>= 1) {
    if (t < o) { s1[t] += s1[t + o]; s2[t] += s2[t + o]; }
    __syncthreads();
  }
  const float impS = s1[0], ldS = s2[0];
  __syncthreads();
  s1[t] = (imp[t] / (impS + 1e-9f)) * (ld[t] / (ldS + 1e-9f));
  __syncthreads();
  for (int o = 64; o >= 1; o >>= 1) {
    if (t < o) s1[t] += s1[t + o];
    __syncthreads();
  }
  if (t == 0) out[0] = (float)NEXP * s1[0];
}

extern "C" void kernel_launch(void* const* d_in, const int* in_sizes, int n_in,
                              void* d_out, int out_size, void* d_ws, size_t ws_size,
                              hipStream_t stream) {
  (void)in_sizes; (void)n_in; (void)out_size; (void)ws_size;
  const float* node = (const float*)d_in[0];
  const float* fea  = (const float*)d_in[1];
  const float* rw   = (const float*)d_in[2];
  const float* w1   = (const float*)d_in[3];
  const float* b1   = (const float*)d_in[4];
  const float* w2   = (const float*)d_in[5];
  const float* b2   = (const float*)d_in[6];
  float* out = (float*)d_out;

  const size_t YELEMS = (size_t)B_TOK * ROWS * CDIM;
  const size_t WELEMS = (size_t)NEXP * HDIM * CDIM;

  char* ws = (char*)d_ws;
  size_t off = 0;
  auto alloc = [&](size_t bytes) -> void* {
    void* p = ws + off;
    off += (bytes + 255) & ~(size_t)255;
    return p;
  };
  __bf16* w1t      = (__bf16*)alloc(WELEMS * sizeof(__bf16));
  __bf16* w2t      = (__bf16*)alloc(WELEMS * sizeof(__bf16));
  float*  topk_w   = (float*)alloc((size_t)B_TOK * TOPK * sizeof(float));
  int*    flat_exp = (int*)alloc((size_t)B_TOK * TOPK * sizeof(int));
  int*    rank     = (int*)alloc((size_t)B_TOK * TOPK * sizeof(int));
  int*    tok_slot = (int*)alloc((size_t)NEXP * CAPN * sizeof(int));
  float*  w_slot   = (float*)alloc((size_t)NEXP * CAPN * sizeof(float));
  float*  imp      = (float*)alloc(NEXP * sizeof(float));
  float*  loadv    = (float*)alloc(NEXP * sizeof(float));

  hipMemsetAsync(out, 0, YELEMS * sizeof(float), stream);
  hipMemsetAsync(tok_slot, 0, (size_t)NEXP * CAPN * sizeof(int), stream);
  hipMemsetAsync(w_slot, 0, (size_t)NEXP * CAPN * sizeof(float), stream);
  hipMemsetAsync(imp, 0, NEXP * sizeof(float), stream);

  convert_w1<<<(int)(WELEMS / 256), 256, 0, stream>>>(w1, w1t);
  convert_w2<<<(int)(WELEMS / 256), 256, 0, stream>>>(w2, w2t);
  router_kernel<<<B_TOK, 128, 0, stream>>>(fea, rw, topk_w, flat_exp, imp);
  rank_kernel<<<NEXP, 256, 0, stream>>>(topk_w, flat_exp, rank, tok_slot, loadv);
  renorm_kernel<<<(B_TOK + 255) / 256, 256, 0, stream>>>(topk_w, flat_exp, rank, w_slot);
  moe_expert_kernel<<<NEXP * CAPN, 256, 0, stream>>>(node, tok_slot, w_slot,
                                                     w1t, w2t, b1, b2, out);
  aux_kernel<<<1, 128, 0, stream>>>(imp, loadv, out + YELEMS);
}
